// CausalSelfAttention_16071767622362
// MI455X (gfx1250) — matrix operand order repaired and checked by reading, not compiled, not run on hardware
//
#include <hip/hip_runtime.h>
#include <hip/hip_bf16.h>
#include <math.h>

// ---------------------------------------------------------------------------
// CausalSelfAttention forward for MI455X (gfx1250, wave32, WMMA).
// B=4, T=2048, D=1024, H=16, Dh=64.  All matmuls in bf16 WMMA w/ f32 accum.
// Data movement: async global->LDS copies (ASYNCcnt) and TDM tensor loads
// (TENSORcnt) when the toolchain exposes them; VGPR-roundtrip fallback else.
// ---------------------------------------------------------------------------

typedef __attribute__((ext_vector_type(16))) __bf16 v16bf;
typedef __attribute__((ext_vector_type(8)))  __bf16 v8bf;
typedef __attribute__((ext_vector_type(8)))  float  v8f;
typedef __attribute__((ext_vector_type(4)))  unsigned int v4u;
typedef __attribute__((ext_vector_type(8)))  int v8i;
typedef __attribute__((ext_vector_type(4)))  int v4i;

#define TSEQ   2048
#define DMODEL 1024
#define NHEAD  16
#define DHEAD  64
#define MTOT   8192          // B*T
#define N_QKV  3072          // 3*D

// ---- CDNA5 feature gates (device pass only; host pass parses fallback) ----
#if defined(__gfx1250__) && __has_builtin(__builtin_amdgcn_global_load_async_to_lds_b128) && __has_builtin(__builtin_amdgcn_s_wait_asynccnt)
#define USE_ASYNC_LDS 1
#else
#define USE_ASYNC_LDS 0
#endif

#if defined(__gfx1250__) && __has_builtin(__builtin_amdgcn_tensor_load_to_lds) && __has_builtin(__builtin_amdgcn_s_wait_tensorcnt)
#define USE_TDM 1
#else
#define USE_TDM 0
#endif

#if __has_include(<hip/amd_detail/amd_gfx1250_TDM.h>)
#define TDM_6ARG 1
#else
#define TDM_6ARG 0
#endif

typedef __attribute__((address_space(3))) void lds_void;
typedef __attribute__((address_space(1))) v4i glob_v4i;   // global 128b chunk
typedef __attribute__((address_space(3))) v4i lds_v4i;    // LDS    128b chunk

__device__ __forceinline__ lds_void* to_lds(void* p) {
  return (lds_void*)p;                       // generic -> LDS addrspacecast
}

// One 16B global->LDS chunk copy (async engine if available).
__device__ __forceinline__ void copy16_g2l(const __bf16* g, __bf16* s) {
#if USE_ASYNC_LDS
  glob_v4i* gp = (glob_v4i*)(unsigned long long)(uintptr_t)g;  // flat==global
  lds_v4i*  lp = (lds_v4i*)to_lds((void*)s);
  __builtin_amdgcn_global_load_async_to_lds_b128(gp, lp, 0, 0);
#else
  *(v8bf*)s = *(const v8bf*)g;
#endif
}

// Fence for staged LDS tiles: drain this wave's async copies, then barrier.
__device__ __forceinline__ void lds_copy_fence() {
#if USE_ASYNC_LDS
  __builtin_amdgcn_s_wait_asynccnt(0);
#endif
  __syncthreads();
}

#if USE_TDM
// TDM 2D tile load: tile0 x tile1 elements of 2B data, row stride = stride0.
// D# group0: [1:0]=count=1, [63:32]=lds_addr, [120:64]=global_addr, [127:126]=type 2.
// D# group1: [17:16]=data_size(1=2B), [79:48]=tensor_dim0, [111:80]=tensor_dim1,
//            [127:112]=tile_dim0, [143:128]=tile_dim1, [207:160]=tensor_dim0_stride.
__device__ __forceinline__ void tdm_load_2d(const void* gptr, unsigned lds_off,
                                            unsigned dim0, unsigned dim1,
                                            unsigned tile0, unsigned tile1,
                                            unsigned stride0) {
  unsigned long long ga = (unsigned long long)(uintptr_t)gptr;
  v4u g0;
  g0[0] = 1u;                                          // count=1
  g0[1] = lds_off;                                     // lds_addr
  g0[2] = (unsigned)(ga & 0xffffffffu);                // global_addr lo
  g0[3] = (unsigned)((ga >> 32) & 0x01ffffffu) | (2u << 30);  // addr hi | type=2
  v8i g1;
  g1[0] = (int)(1u << 16);                             // data_size = 2 bytes
  g1[1] = (int)((dim0 & 0xffffu) << 16);               // tensor_dim0[15:0]
  g1[2] = (int)(((dim0 >> 16) & 0xffffu) | ((dim1 & 0xffffu) << 16));
  g1[3] = (int)(((dim1 >> 16) & 0xffffu) | ((tile0 & 0xffffu) << 16));
  g1[4] = (int)(tile1 & 0xffffu);                      // tile_dim1 (tile_dim2=0)
  g1[5] = (int)stride0;                                // tensor_dim0_stride lo32
  g1[6] = 0;
  g1[7] = 0;
  v4i z4; z4[0] = 0; z4[1] = 0; z4[2] = 0; z4[3] = 0;
#if TDM_6ARG
  v8i z8; for (int i = 0; i < 8; ++i) z8[i] = 0;
  __builtin_amdgcn_tensor_load_to_lds(g0, g1, z4, z4, z8, 0);
#else
  __builtin_amdgcn_tensor_load_to_lds(g0, g1, z4, z4, 0);
#endif
}
#endif  // USE_TDM

__device__ __forceinline__ v8f wmma_bf16(v16bf a, v16bf b, v8f c) {
  return __builtin_amdgcn_wmma_f32_16x16x32_bf16(
      /*neg_a=*/false, a, /*neg_b=*/false, b,
      /*c_mod=*/(short)0, c, /*reuse_a=*/false, /*reuse_b=*/false);
}

// A-matrix fragment (16x32, MxK).  ISA 7.12.2: lanes 0-15 row M=lane,
// VGPR0..3 = K 0..7 (grp0) / 8..15 (grp1), VGPR4..7 = K 16..23 / 24..31.
__device__ __forceinline__ v16bf load_frag_a(const __bf16* base, int stride) {
  int lane = threadIdx.x & 31;
  int r = lane & 15, grp = lane >> 4;
  const __bf16* p = base + r * stride + grp * 8;
  union { v16bf v; v8bf h[2]; } u;
  u.h[0] = *(const v8bf*)(p);        // K = grp*8 .. grp*8+7
  u.h[1] = *(const v8bf*)(p + 16);   // K = 16+grp*8 .. +7
  return u.v;
}

// B-matrix fragment (32x16, KxN) from an N-major LDS tile (row = N, col = K).
__device__ __forceinline__ v16bf load_frag_b(const __bf16* base, int stride) {
  int lane = threadIdx.x & 31;
  int r = lane & 15, grp = lane >> 4;
  const __bf16* p = base + r * stride + grp * 8;
  union { v16bf v; v8bf h[2]; } u;
  u.h[0] = *(const v8bf*)(p);        // K = grp*8 .. grp*8+7
  u.h[1] = *(const v8bf*)(p + 16);   // K = 16+grp*8 .. +7
  return u.v;
}

// 128x64 bf16 tile, global (already offset to row0/k0, leading dim ldg) -> LDS.
__device__ __forceinline__ void load_tile_128x64(const __bf16* __restrict__ g,
                                                 long ldg, __bf16* s) {
  int tid = threadIdx.x;
#pragma unroll
  for (int i = 0; i < 4; ++i) {
    int c = tid + i * 256;           // 0..1023 16B chunks
    int r = c >> 3;                  // 0..127
    int o = (c & 7) * 8;             // 0..56
    copy16_g2l(g + (long)r * ldg + o, s + r * 64 + o);
  }
}

// ---------------------------------------------------------------------------
// Elementwise converters
// ---------------------------------------------------------------------------
__global__ __launch_bounds__(256) void f32_to_bf16_kernel(
    const float* __restrict__ src, __bf16* __restrict__ dst, int n) {
  int i = blockIdx.x * 256 + threadIdx.x;
  if (i < n) dst[i] = (__bf16)src[i];
}

// src [R][C] f32 -> dst [C][R] bf16
__global__ __launch_bounds__(256) void transpose_f32_to_bf16_kernel(
    const float* __restrict__ src, __bf16* __restrict__ dst, int R, int C) {
  int i = blockIdx.x * 256 + threadIdx.x;
  if (i < R * C) {
    int r = i / C, c = i - r * C;
    dst[(long)c * R + r] = (__bf16)src[i];
  }
}

// ---------------------------------------------------------------------------
// Shared GEMM core: acc(4x2 16x16 tiles per wave) over K=DMODEL
// ---------------------------------------------------------------------------
#define GEMM_CORE(APTR, BPTR)                                                  \
  const int wave = threadIdx.x >> 5;                                           \
  const int wm = wave >> 2, wn = wave & 3;                                     \
  const int lane = threadIdx.x & 31, nl = lane & 15, hi = lane >> 4;           \
  v8f acc[4][2];                                                               \
  _Pragma("unroll") for (int a_ = 0; a_ < 4; ++a_)                             \
  _Pragma("unroll") for (int b_ = 0; b_ < 2; ++b_)                             \
  _Pragma("unroll") for (int e_ = 0; e_ < 8; ++e_) acc[a_][b_][e_] = 0.f;      \
  const int nK = DMODEL / 64;                                                  \
  load_tile_128x64((APTR) + (long)m0 * DMODEL, DMODEL, As);                    \
  load_tile_128x64((BPTR) + (long)n0 * DMODEL, DMODEL, Bs);                    \
  lds_copy_fence();                                                            \
  for (int kt = 0; kt < nK; ++kt) {                                            \
    int cur = kt & 1;                                                          \
    if (kt + 1 < nK) {                                                         \
      load_tile_128x64((APTR) + (long)m0 * DMODEL + (kt + 1) * 64, DMODEL,     \
                       As + (cur ^ 1) * 8192);                                 \
      load_tile_128x64((BPTR) + (long)n0 * DMODEL + (kt + 1) * 64, DMODEL,     \
                       Bs + (cur ^ 1) * 8192);                                 \
    }                                                                          \
    if (kt + 2 < nK) {                                                         \
      __builtin_prefetch((APTR) + (long)(m0 + (threadIdx.x >> 1)) * DMODEL +   \
                             (kt + 2) * 64, 0, 0);                             \
      __builtin_prefetch((BPTR) + (long)(n0 + (threadIdx.x >> 1)) * DMODEL +   \
                             (kt + 2) * 64, 0, 0);                             \
    }                                                                          \
    const __bf16* Ac = As + cur * 8192;                                        \
    const __bf16* Bc = Bs + cur * 8192;                                        \
    _Pragma("unroll") for (int ks = 0; ks < 64; ks += 32) {                    \
      v16bf af[4];                                                             \
      _Pragma("unroll") for (int mt = 0; mt < 4; ++mt)                         \
        af[mt] = load_frag_a(Ac + (wm * 64 + mt * 16) * 64 + ks, 64);          \
      v16bf bf0 = load_frag_b(Bc + (wn * 32 + 0) * 64 + ks, 64);               \
      v16bf bf1 = load_frag_b(Bc + (wn * 32 + 16) * 64 + ks, 64);              \
      _Pragma("unroll") for (int mt = 0; mt < 4; ++mt) {                       \
        acc[mt][0] = wmma_bf16(af[mt], bf0, acc[mt][0]);                       \
        acc[mt][1] = wmma_bf16(af[mt], bf1, acc[mt][1]);                       \
      }                                                                        \
    }                                                                          \
    lds_copy_fence();                                                          \
  }

// ---------------------------------------------------------------------------
// GEMM 1: [8192,1024]x[1024,3072] -> scatter into Q/K (b,h,t,dh) and Vt (b,h,dh,t)
// ---------------------------------------------------------------------------
__global__ __launch_bounds__(256) void gemm_qkv_kernel(
    const __bf16* __restrict__ Xb, const __bf16* __restrict__ Wt,
    __bf16* __restrict__ Qd, __bf16* __restrict__ Kd, __bf16* __restrict__ Vtd) {
  __shared__ __bf16 As[2 * 128 * 64];
  __shared__ __bf16 Bs[2 * 128 * 64];
  const int m0 = blockIdx.y * 128;
  const int n0 = blockIdx.x * 128;
  GEMM_CORE(Xb, Wt)
#pragma unroll
  for (int mt = 0; mt < 4; ++mt)
#pragma unroll
    for (int nt = 0; nt < 2; ++nt)
#pragma unroll
      for (int g = 0; g < 8; ++g) {
        int m = m0 + wm * 64 + mt * 16 + g + 8 * hi;
        int n = n0 + wn * 32 + nt * 16 + nl;
        float val = acc[mt][nt][g];
        int which = n >> 10, dcol = n & 1023;
        int h = dcol >> 6, dh = dcol & 63;
        int b = m >> 11, t = m & 2047;
        long bh = (long)b * NHEAD + h;
        if (which == 0)
          Qd[(bh * TSEQ + t) * DHEAD + dh] = (__bf16)val;
        else if (which == 1)
          Kd[(bh * TSEQ + t) * DHEAD + dh] = (__bf16)val;
        else
          Vtd[(bh * DHEAD + dh) * TSEQ + t] = (__bf16)val;
      }
}

// ---------------------------------------------------------------------------
// GEMM 2: [8192,1024]x[1024,1024] -> f32 out
// ---------------------------------------------------------------------------
__global__ __launch_bounds__(256) void gemm_proj_kernel(
    const __bf16* __restrict__ Yb, const __bf16* __restrict__ Wt,
    float* __restrict__ out) {
  __shared__ __bf16 As[2 * 128 * 64];
  __shared__ __bf16 Bs[2 * 128 * 64];
  const int m0 = blockIdx.y * 128;
  const int n0 = blockIdx.x * 128;
  GEMM_CORE(Yb, Wt)
#pragma unroll
  for (int mt = 0; mt < 4; ++mt)
#pragma unroll
    for (int nt = 0; nt < 2; ++nt)
#pragma unroll
      for (int g = 0; g < 8; ++g) {
        int m = m0 + wm * 64 + mt * 16 + g + 8 * hi;
        int n = n0 + wn * 32 + nt * 16 + nl;
        out[(long)m * DMODEL + n] = acc[mt][nt][g];
      }
}

// ---------------------------------------------------------------------------
// RoPE in place on Q and K: one thread per (bh, t, pair)
// ---------------------------------------------------------------------------
__global__ __launch_bounds__(256) void rope_kernel(__bf16* __restrict__ Q,
                                                   __bf16* __restrict__ K) {
  int i = blockIdx.x * 256 + threadIdx.x;   // 64*2048*32 total
  int j = i & 31;
  int t = (i >> 5) & 2047;
  int bh = i >> 16;
  long base = ((long)bh * TSEQ + t) * DHEAD + 2 * j;
  float theta = (float)t * __expf(-(float)(2 * j) * (9.2103403719761836f / 64.f));
  float sn, cs;
  __sincosf(theta, &sn, &cs);
  float q0 = (float)Q[base], q1 = (float)Q[base + 1];
  Q[base]     = (__bf16)(q0 * cs - q1 * sn);
  Q[base + 1] = (__bf16)(q1 * cs + q0 * sn);
  float k0 = (float)K[base], k1 = (float)K[base + 1];
  K[base]     = (__bf16)(k0 * cs - k1 * sn);
  K[base + 1] = (__bf16)(k1 * cs + k0 * sn);
}

// ---------------------------------------------------------------------------
// Flash attention: block = (b,h) x 128 q rows; 8 waves each own 16 q rows.
// K tile staged via TDM (wave 0 issues, TENSORcnt), V via async copies.
// ---------------------------------------------------------------------------
__global__ __launch_bounds__(256) void attn_kernel(
    const __bf16* __restrict__ Q, const __bf16* __restrict__ Kg,
    const __bf16* __restrict__ Vt, __bf16* __restrict__ Y) {
  __shared__ __bf16 Qs[128 * 64];     // q rows x dh
  __shared__ __bf16 Ks[128 * 64];     // key rows x dh (N-major for B of QK^T)
  __shared__ __bf16 Vs[64 * 128];     // dh x keys    (N-major for B of P.V)
  __shared__ __bf16 Ps[8 * 16 * 128]; // per-wave P tile

  const int bh = blockIdx.y;
  const int qt = blockIdx.x;
  const __bf16* Qp = Q  + (long)bh * TSEQ * DHEAD + (long)qt * 128 * DHEAD;
  const __bf16* Kp = Kg + (long)bh * TSEQ * DHEAD;
  const __bf16* Vp = Vt + (long)bh * DHEAD * TSEQ;

  const int tid = threadIdx.x;
  const int wave = tid >> 5, lane = tid & 31, nl = lane & 15, hi = lane >> 4;

  // load the Q tile once
#pragma unroll
  for (int i = 0; i < 4; ++i) {
    int c = tid + i * 256, r = c >> 3, o = (c & 7) * 8;
    copy16_g2l(Qp + (long)r * 64 + o, Qs + r * 64 + o);
  }

  v8f oacc[4];
  float mrow[8], lrow[8];
#pragma unroll
  for (int j = 0; j < 4; ++j)
#pragma unroll
    for (int e = 0; e < 8; ++e) oacc[j][e] = 0.f;
#pragma unroll
  for (int g = 0; g < 8; ++g) { mrow[g] = -__builtin_inff(); lrow[g] = 0.f; }

  lds_copy_fence();

  const float sc = 0.125f * 1.44269504088896f;  // 1/sqrt(64) * log2(e)

  for (int kt = 0; kt <= qt; ++kt) {
    // stage K tile [128 keys][64] ...
#if USE_TDM
    if (wave == 0) {
      unsigned ks_off = (unsigned)(uintptr_t)to_lds((void*)Ks);
      tdm_load_2d(Kp + (long)kt * 128 * DHEAD, ks_off,
                  /*dim0=*/DHEAD, /*dim1=*/(unsigned)(TSEQ - kt * 128),
                  /*tile0=*/DHEAD, /*tile1=*/128, /*stride0=*/DHEAD);
      __builtin_amdgcn_s_wait_tensorcnt(0);
    }
#else
#pragma unroll
    for (int i = 0; i < 4; ++i) {
      int c = tid + i * 256, r = c >> 3, o = (c & 7) * 8;
      copy16_g2l(Kp + (long)(kt * 128 + r) * 64 + o, Ks + r * 64 + o);
    }
#endif
    // ... and V tile [64 dh][128 keys]
#pragma unroll
    for (int i = 0; i < 4; ++i) {
      int c = tid + i * 256, r = c >> 4, o = (c & 15) * 8;
      copy16_g2l(Vp + (long)r * TSEQ + kt * 128 + o, Vs + r * 128 + o);
    }
    lds_copy_fence();

    // S = Q K^T  (wave's 16 rows x 128 cols)
    v8f sacc[8];
#pragma unroll
    for (int nt = 0; nt < 8; ++nt)
#pragma unroll
      for (int e = 0; e < 8; ++e) sacc[nt][e] = 0.f;
#pragma unroll
    for (int ks = 0; ks < 64; ks += 32) {
      v16bf qf = load_frag_a(Qs + (wave * 16) * 64 + ks, 64);
#pragma unroll
      for (int nt = 0; nt < 8; ++nt) {
        v16bf kf = load_frag_b(Ks + (nt * 16) * 64 + ks, 64);
        sacc[nt] = wmma_bf16(qf, kf, sacc[nt]);
      }
    }

    // online softmax (exp2 domain); causal mask; stash P as bf16 in LDS
#pragma unroll
    for (int g = 0; g < 8; ++g) {
      int m_loc = g + 8 * hi;
      int mg = qt * 128 + wave * 16 + m_loc;
      float sv[8];
      float rmax = -__builtin_inff();
#pragma unroll
      for (int nt = 0; nt < 8; ++nt) {
        int ng = kt * 128 + nt * 16 + nl;
        float s = sacc[nt][g] * sc;
        if (ng > mg) s = -__builtin_inff();
        sv[nt] = s;
        rmax = fmaxf(rmax, s);
      }
#pragma unroll
      for (int d = 1; d < 16; d <<= 1) rmax = fmaxf(rmax, __shfl_xor(rmax, d, 32));
      float newm = fmaxf(mrow[g], rmax);
      float alpha = exp2f(mrow[g] - newm);
      float rsum = 0.f;
#pragma unroll
      for (int nt = 0; nt < 8; ++nt) {
        float p = exp2f(sv[nt] - newm);
        rsum += p;
        Ps[wave * 2048 + m_loc * 128 + nt * 16 + nl] = (__bf16)p;
      }
#pragma unroll
      for (int d = 1; d < 16; d <<= 1) rsum += __shfl_xor(rsum, d, 32);
      lrow[g] = lrow[g] * alpha + rsum;
      mrow[g] = newm;
#pragma unroll
      for (int j = 0; j < 4; ++j) oacc[j][g] *= alpha;
    }

    // O += P V   (wave-private P in LDS: DS ops are in-order per wave)
#pragma unroll
    for (int ks2 = 0; ks2 < 4; ++ks2) {
      v16bf pf = load_frag_a(Ps + wave * 2048 + ks2 * 32, 128);
#pragma unroll
      for (int j = 0; j < 4; ++j) {
        v16bf vf = load_frag_b(Vs + (j * 16) * 128 + ks2 * 32, 128);
        oacc[j] = wmma_bf16(pf, vf, oacc[j]);
      }
    }
    __syncthreads();
  }

  // normalize + store Y [B*T][D] bf16
  int b = bh >> 4, h = bh & 15;
#pragma unroll
  for (int j = 0; j < 4; ++j)
#pragma unroll
    for (int g = 0; g < 8; ++g) {
      int m_loc = g + 8 * hi;
      int tg = qt * 128 + wave * 16 + m_loc;
      int dh = j * 16 + nl;
      float v = oacc[j][g] / lrow[g];
      Y[((long)(b * TSEQ + tg)) * DMODEL + h * DHEAD + dh] = (__bf16)v;
    }
}

// ---------------------------------------------------------------------------
// Host launcher
// ---------------------------------------------------------------------------
extern "C" void kernel_launch(void* const* d_in, const int* in_sizes, int n_in,
                              void* d_out, int out_size, void* d_ws, size_t ws_size,
                              hipStream_t stream) {
  (void)in_sizes; (void)n_in; (void)out_size; (void)ws_size;
  const float* x  = (const float*)d_in[0];
  const float* Wa = (const float*)d_in[1];
  const float* Wp = (const float*)d_in[2];
  float* out = (float*)d_out;

  char* ws = (char*)d_ws;
  size_t off = 0;
  __bf16* Xb  = (__bf16*)(ws + off); off += (size_t)MTOT * DMODEL * 2;        // 16 MiB
  __bf16* Wat = (__bf16*)(ws + off); off += (size_t)N_QKV * DMODEL * 2;       //  6 MiB
  __bf16* Wpt = (__bf16*)(ws + off); off += (size_t)DMODEL * DMODEL * 2;      //  2 MiB
  __bf16* Qd  = (__bf16*)(ws + off); off += (size_t)64 * TSEQ * DHEAD * 2;    // 16 MiB
  __bf16* Kd  = (__bf16*)(ws + off); off += (size_t)64 * TSEQ * DHEAD * 2;    // 16 MiB
  __bf16* Vtd = (__bf16*)(ws + off); off += (size_t)64 * DHEAD * TSEQ * 2;    // 16 MiB
  __bf16* Yb  = (__bf16*)(ws + off); off += (size_t)MTOT * DMODEL * 2;        // 16 MiB

  f32_to_bf16_kernel<<<(MTOT * DMODEL) / 256, 256, 0, stream>>>(x, Xb, MTOT * DMODEL);
  transpose_f32_to_bf16_kernel<<<(DMODEL * N_QKV) / 256, 256, 0, stream>>>(Wa, Wat, DMODEL, N_QKV);
  transpose_f32_to_bf16_kernel<<<(DMODEL * DMODEL) / 256, 256, 0, stream>>>(Wp, Wpt, DMODEL, DMODEL);

  gemm_qkv_kernel<<<dim3(N_QKV / 128, MTOT / 128), 256, 0, stream>>>(Xb, Wat, Qd, Kd, Vtd);
  rope_kernel<<<(64 * TSEQ * 32) / 256, 256, 0, stream>>>(Qd, Kd);
  attn_kernel<<<dim3(TSEQ / 128, 64), 256, 0, stream>>>(Qd, Kd, Vtd, Yb);
  gemm_proj_kernel<<<dim3(DMODEL / 128, MTOT / 128), 256, 0, stream>>>(Yb, Wpt, out);
}
